// Model_51402168599091
// MI455X (gfx1250) — matrix operand order repaired and checked by reading, not compiled, not run on hardware
//
#include <hip/hip_runtime.h>
#include <hip/hip_bf16.h>

typedef _Float16 half_t;
typedef _Float16 v16h __attribute__((ext_vector_type(16)));
typedef float    v8f  __attribute__((ext_vector_type(8)));

#define EMB   256
#define NTOK  394
#define NKPAD 416   // keys padded to 26*16 (13 k-steps of 32)

__device__ __forceinline__ v8f wmma16(v16h a, v16h b, v8f c) {
  // D = A(16x32 f16) x B(32x16 f16) + C(16x16 f32)
  return __builtin_amdgcn_wmma_f32_16x16x32_f16(false, a, false, b, (short)0, c, false, false);
}

// Identity chunk mapping within each 32-element K block: lane-group g's A
// fragment (K = g*16..g*16+15) is contiguous at offset g*16.
__device__ __forceinline__ int perm32(int e) {
  int off = e & 31;
  int chunk = off >> 3;                       // 0..3
  int nc = chunk;                             // identity
  return (e & ~31) | (nc << 3) | (off & 7);
}

// ---------------- pack weight (K x 256 fp32) into WMMA B-operand order ----------
// dst[(((ks*2 + g)*256 + col)*16 + j] = W[(ks*32 + g*16 + j)*256 + col]
__global__ void k_pack_w(const float* __restrict__ W, half_t* __restrict__ dst, int n) {
  int i = blockIdx.x * 256 + threadIdx.x;
  if (i >= n) return;
  int j   = i & 15;
  int col = (i >> 4) & 255;
  int gsk = i >> 12;
  int g   = gsk & 1;
  int ks  = gsk >> 1;
  dst[i] = (half_t)W[(size_t)(ks * 32 + g * 16 + j) * EMB + col];
}

__global__ void k_write_cls(const float* __restrict__ cls1, const float* __restrict__ cls2,
                            float* __restrict__ z) {
  int b = blockIdx.x, e = threadIdx.x;
  z[((size_t)b * NTOK + 0)   * EMB + e] = cls2[e];   // token 0   = cls2 (p2r[0])
  z[((size_t)b * NTOK + 197) * EMB + e] = cls1[e];   // token 197 = cls1
}

// ---------------- patch-embed GEMM (gather-fused, WMMA) ----------------
__global__ __launch_bounds__(128) void k_patch_gemm(
    const float* __restrict__ x, const half_t* __restrict__ Wp,
    const float* __restrict__ bias, float* __restrict__ z,
    int K, int p, int wg, int pid_mul, int pid_add, int token_add)
{
  __shared__ alignas(32) half_t As[16 * 32];
  const int tid = threadIdx.x;
  const int lane = tid & 31, wave = tid >> 5;
  const int g = lane >> 4, ml = lane & 15;
  const int rowbase = blockIdx.x * 16;
  v8f acc[4] = {};
  // hoist per-staged-element geometry (idx = tid + u*128 fixed per thread)
  int pm[4], pb[4], pph[4], ppw[4];
#pragma unroll
  for (int u = 0; u < 4; ++u) {
    int idx = tid + u * 128;
    int m = idx >> 5;
    int row = rowbase + m;
    int b = row / 196, r = row - b * 196;
    int pid = r * pid_mul + pid_add;
    pm[u] = perm32(idx) & 511;  // permuted LDS slot (same m, permuted kk)
    pb[u] = b;
    pph[u] = pid / wg;
    ppw[u] = pid - pph[u] * wg;
  }
  const int ksteps = K >> 5;
  for (int ks = 0; ks < ksteps; ++ks) {
    __syncthreads();
#pragma unroll
    for (int u = 0; u < 4; ++u) {
      int idx = tid + u * 128;
      int kk = idx & 31;
      int k = (ks << 5) + kk;
      int c = k % 3, qq = k / 3;
      int px = qq % p, py = qq / p;
      As[pm[u]] = (half_t)x[(((size_t)pb[u] * 3 + c) * 224 + (pph[u] * p + py)) * 224 +
                            (ppw[u] * p + px)];
    }
    __syncthreads();
    v16h a = *reinterpret_cast<const v16h*>(&As[ml * 32 + g * 16]);
#pragma unroll
    for (int nt = 0; nt < 4; ++nt) {
      int col = wave * 64 + nt * 16 + ml;
      v16h bt = *reinterpret_cast<const v16h*>(Wp + ((size_t)((ks * 2 + g) * 256 + col) << 4));
      acc[nt] = wmma16(a, bt, acc[nt]);
    }
  }
#pragma unroll
  for (int nt = 0; nt < 4; ++nt) {
    int col = wave * 64 + nt * 16 + ml;
    float bv = bias[col];
#pragma unroll
    for (int i = 0; i < 8; ++i) {
      int row = rowbase + i + g * 8;
      int b = row / 196, r = row - b * 196;
      z[((size_t)b * NTOK + (r + token_add)) * EMB + col] = acc[nt][i] + bv;
    }
  }
}

// ---------------- LN1 -> f16 (A-permuted layout) ----------------
__global__ __launch_bounds__(256) void k_ln_f16(const float* __restrict__ z,
                                                const float* __restrict__ gg,
                                                const float* __restrict__ bb,
                                                half_t* __restrict__ h) {
  __shared__ float red[256];
  int t = blockIdx.x, e = threadIdx.x;
  float v = z[(size_t)t * EMB + e];
  red[e] = v; __syncthreads();
  for (int s = 128; s > 0; s >>= 1) { if (e < s) red[e] += red[e + s]; __syncthreads(); }
  float mean = red[0] * (1.0f / EMB);
  __syncthreads();
  float d = v - mean;
  red[e] = d * d; __syncthreads();
  for (int s = 128; s > 0; s >>= 1) { if (e < s) red[e] += red[e + s]; __syncthreads(); }
  float rs = rsqrtf(red[0] * (1.0f / EMB) + 1e-5f);
  h[(size_t)t * EMB + perm32(e)] = (half_t)(d * rs * gg[e] + bb[e]);
}

// ---------------- QKV GEMM: out(f16) = A(f16,Aperm) @ Wpacked + bias --------------
// mode 0: plain row-major (K matrix; contiguous-in-d B loads in attention)
// mode 1: A-permuted   (Q matrix; consumed as WMMA A operand)
// mode 2: V B-packed   ([bh][ks][g][d][j] for the P@V GEMM)
__global__ __launch_bounds__(128) void k_gemm_qkv(const half_t* __restrict__ A,
                                                  const half_t* __restrict__ Wp,
                                                  const float* __restrict__ bias,
                                                  half_t* __restrict__ out, int mode) {
  const int tid = threadIdx.x, lane = tid & 31, wave = tid >> 5;
  const int g = lane >> 4, ml = lane & 15;
  const int rowbase = blockIdx.x * 16;
  v8f acc[4] = {};
  const half_t* arow = A + (size_t)(rowbase + ml) * EMB;
  for (int ks = 0; ks < 8; ++ks) {
    v16h a = *reinterpret_cast<const v16h*>(arow + ks * 32 + g * 16);
#pragma unroll
    for (int nt = 0; nt < 4; ++nt) {
      int col = wave * 64 + nt * 16 + ml;
      v16h bt = *reinterpret_cast<const v16h*>(Wp + ((size_t)((ks * 2 + g) * 256 + col) << 4));
      acc[nt] = wmma16(a, bt, acc[nt]);
    }
  }
#pragma unroll
  for (int nt = 0; nt < 4; ++nt) {
    int col = wave * 64 + nt * 16 + ml;
    float bv = bias[col];
    int pcol = perm32(col);
    int head = col >> 7, d = col & 127;
#pragma unroll
    for (int i = 0; i < 8; ++i) {
      int row = rowbase + i + g * 8;
      half_t val = (half_t)(acc[nt][i] + bv);
      if (mode == 0) {
        out[(size_t)row * EMB + col] = val;
      } else if (mode == 1) {
        out[(size_t)row * EMB + pcol] = val;
      } else {
        int b = row / NTOK, t = row - b * NTOK;
        int ts = t >> 5, gg2 = (t >> 4) & 1, jj = t & 15;
        out[((((size_t)(b * 2 + head) * 13 + ts) * 2 + gg2) * 128 + d) * 16 + jj] = val;
      }
    }
  }
}

// ---------------- attention: one wave per (b, head, 16-query tile) ----------------
__global__ __launch_bounds__(32) void k_attn(const half_t* __restrict__ q,
                                             const half_t* __restrict__ kk,
                                             const half_t* __restrict__ vp,
                                             half_t* __restrict__ o) {
  __shared__ float  sS[16 * NKPAD];
  __shared__ alignas(32) half_t sP[16 * NKPAD];
  __shared__ float  sInv[16];
  const int lane = threadIdx.x, g = lane >> 4, ml = lane & 15;
  int idx = blockIdx.x;
  const int qt = idx % 25; const int bh = idx / 25;
  const int head = bh & 1; const int b = bh >> 1;
  const int q0 = qt * 16;

  int tq = q0 + ml; if (tq > NTOK - 1) tq = NTOK - 1;
  const half_t* qrow = q + ((size_t)(b * NTOK + tq) * EMB + head * 128);  // A-permuted
  v16h aq[4];
#pragma unroll
  for (int ks = 0; ks < 4; ++ks)
    aq[ks] = *reinterpret_cast<const v16h*>(qrow + ks * 32 + g * 16);

  // scores S = q @ k^T  (keys padded to 416, masked to -inf)
  for (int kt = 0; kt < 26; ++kt) {
    int tk = kt * 16 + ml; int tkc = tk > NTOK - 1 ? NTOK - 1 : tk;
    const half_t* krow = kk + ((size_t)(b * NTOK + tkc) * EMB + head * 128); // plain
    v8f c = {};
#pragma unroll
    for (int ks = 0; ks < 4; ++ks) {
      v16h bt = *reinterpret_cast<const v16h*>(krow + ks * 32 + g * 16);
      c = wmma16(aq[ks], bt, c);
    }
    int col = kt * 16 + ml;
#pragma unroll
    for (int i = 0; i < 8; ++i) {
      int m = i + g * 8;
      sS[m * NKPAD + col] = (col < NTOK) ? c[i] : -1e30f;
    }
  }

  // row softmax (2 lanes per row); /16 folded into sInv; P stored A-permuted
  {
    const int r = ml; const int base = r * NKPAD + g * (NKPAD / 2);
    float mx = -1e30f;
    for (int cj = 0; cj < NKPAD / 2; ++cj) mx = fmaxf(mx, sS[base + cj]);
    mx = fmaxf(mx, __shfl_xor(mx, 16, 32));
    float sum = 0.f;
    for (int cj = 0; cj < NKPAD / 2; ++cj) {
      int colg = g * (NKPAD / 2) + cj;
      float ev = __expf(sS[r * NKPAD + colg] - mx);
      sP[r * NKPAD + perm32(colg)] = (half_t)ev;
      sum += ev;
    }
    sum += __shfl_xor(sum, 16, 32);
    if (g == 0) sInv[r] = 1.0f / (sum * 16.0f);
  }

  // O = P @ V  (K = 416 padded, 8 n-tiles of 16 over dh=128); V is B-packed
  for (int nt = 0; nt < 8; ++nt) {
    v8f c = {};
    int dcol = nt * 16 + ml;
    for (int ks = 0; ks < 13; ++ks) {
      v16h a = *reinterpret_cast<const v16h*>(&sP[ml * NKPAD + ks * 32 + g * 16]);
      v16h bt = *reinterpret_cast<const v16h*>(
          vp + ((((size_t)(b * 2 + head) * 13 + ks) * 2 + g) * 128 + dcol) * 16);
      c = wmma16(a, bt, c);
    }
    int pcol = perm32(head * 128 + dcol);   // store A-permuted for the proj GEMM
#pragma unroll
    for (int i = 0; i < 8; ++i) {
      int m = i + g * 8; int t = q0 + m;
      if (t < NTOK)
        o[(size_t)(b * NTOK + t) * EMB + pcol] = (half_t)(c[i] * sInv[m]);
    }
  }
}

// ---------------- proj GEMM + LN2 + residual add into z ----------------
__global__ __launch_bounds__(128) void k_proj_ln_add(const half_t* __restrict__ A,
                                                     const half_t* __restrict__ Wp,
                                                     const float* __restrict__ bias,
                                                     const float* __restrict__ g2,
                                                     const float* __restrict__ b2,
                                                     float* __restrict__ z) {
  __shared__ float sT[16 * EMB];
  __shared__ float sMean[16], sRstd[16];
  const int tid = threadIdx.x, lane = tid & 31, wave = tid >> 5;
  const int g = lane >> 4, ml = lane & 15;
  const int rowbase = blockIdx.x * 16;
  v8f acc[4] = {};
  const half_t* arow = A + (size_t)(rowbase + ml) * EMB;   // A-permuted
  for (int ks = 0; ks < 8; ++ks) {
    v16h a = *reinterpret_cast<const v16h*>(arow + ks * 32 + g * 16);
#pragma unroll
    for (int nt = 0; nt < 4; ++nt) {
      int col = wave * 64 + nt * 16 + ml;
      v16h bt = *reinterpret_cast<const v16h*>(Wp + ((size_t)((ks * 2 + g) * 256 + col) << 4));
      acc[nt] = wmma16(a, bt, acc[nt]);
    }
  }
#pragma unroll
  for (int nt = 0; nt < 4; ++nt) {
    int col = wave * 64 + nt * 16 + ml;
    float bv = bias[col];
#pragma unroll
    for (int i = 0; i < 8; ++i) sT[(i + g * 8) * EMB + col] = acc[nt][i] + bv;
  }
  __syncthreads();
  if (tid < 16) {
    float s = 0.f, s2 = 0.f;
    for (int e = 0; e < EMB; ++e) { float v = sT[tid * EMB + e]; s += v; s2 += v * v; }
    float mean = s * (1.0f / EMB);
    sMean[tid] = mean;
    sRstd[tid] = rsqrtf(s2 * (1.0f / EMB) - mean * mean + 1e-5f);
  }
  __syncthreads();
  for (int i2 = tid; i2 < 16 * EMB; i2 += 128) {
    int r = i2 >> 8, e = i2 & 255;
    float val = (sT[i2] - sMean[r]) * sRstd[r] * g2[e] + b2[e];
    z[(size_t)(rowbase + r) * EMB + e] += val;
  }
}

// ---------------- pool + LN + classifier head + log_softmax ----------------
__global__ __launch_bounds__(256) void k_head(const float* __restrict__ z,
                                              const float* __restrict__ gg,
                                              const float* __restrict__ bb,
                                              const float* __restrict__ Wc,
                                              const float* __restrict__ bc,
                                              float* __restrict__ out) {
  __shared__ float red[256];
  int b = blockIdx.x, e = threadIdx.x;
  float acc = 0.f;
  for (int t = 0; t < NTOK; ++t) acc += z[((size_t)b * NTOK + t) * EMB + e];
  float pooled = acc * (1.0f / NTOK);
  red[e] = pooled; __syncthreads();
  for (int s = 128; s > 0; s >>= 1) { if (e < s) red[e] += red[e + s]; __syncthreads(); }
  float mean = red[0] * (1.0f / EMB); __syncthreads();
  float d = pooled - mean;
  red[e] = d * d; __syncthreads();
  for (int s = 128; s > 0; s >>= 1) { if (e < s) red[e] += red[e + s]; __syncthreads(); }
  float rs = rsqrtf(red[0] * (1.0f / EMB) + 1e-5f); __syncthreads();
  float nrm = d * rs * gg[e] + bb[e];
  red[e] = nrm * Wc[e * 2 + 0]; __syncthreads();
  for (int s = 128; s > 0; s >>= 1) { if (e < s) red[e] += red[e + s]; __syncthreads(); }
  float l0 = red[0] + bc[0]; __syncthreads();
  red[e] = nrm * Wc[e * 2 + 1]; __syncthreads();
  for (int s = 128; s > 0; s >>= 1) { if (e < s) red[e] += red[e + s]; __syncthreads(); }
  float l1 = red[0] + bc[1];
  if (e == 0) {
    float m = fmaxf(l0, l1);
    float lse = m + __logf(__expf(l0 - m) + __expf(l1 - m));
    out[b * 2 + 0] = l0 - lse;
    out[b * 2 + 1] = l1 - lse;
  }
}

extern "C" void kernel_launch(void* const* d_in, const int* in_sizes, int n_in,
                              void* d_out, int out_size, void* d_ws, size_t ws_size,
                              hipStream_t stream) {
  const float* x     = (const float*)d_in[0];
  const float* W1    = (const float*)d_in[1];
  const float* b1    = (const float*)d_in[2];
  const float* cls1  = (const float*)d_in[3];
  const float* W2    = (const float*)d_in[4];
  const float* b2    = (const float*)d_in[5];
  const float* cls2  = (const float*)d_in[6];
  const float* ln1_g = (const float*)d_in[7];
  const float* ln1_b = (const float*)d_in[8];
  const float* Wq    = (const float*)d_in[9];
  const float* bq    = (const float*)d_in[10];
  const float* Wk    = (const float*)d_in[11];
  const float* bk    = (const float*)d_in[12];
  const float* Wv    = (const float*)d_in[13];
  const float* bv    = (const float*)d_in[14];
  const float* Wp    = (const float*)d_in[15];
  const float* bp    = (const float*)d_in[16];
  const float* ln2_g = (const float*)d_in[17];
  const float* ln2_b = (const float*)d_in[18];
  const float* lnc_g = (const float*)d_in[19];
  const float* lnc_b = (const float*)d_in[20];
  const float* Wc    = (const float*)d_in[21];
  const float* bc    = (const float*)d_in[22];
  float* out = (float*)d_out;

  const int Bn = 128;
  const size_t Mtok = (size_t)Bn * NTOK;            // 50432 tokens
  const size_t VP_HALVES = (size_t)Bn * 2 * 13 * 2 * 128 * 16;  // V B-packed
  char* wsb = (char*)d_ws;
  float*  z   = (float*)wsb;                        // [Bn,394,256] fp32 residual
  half_t* h   = (half_t*)(wsb + Mtok * EMB * 4);    // A-permuted
  half_t* qb  = h  + Mtok * EMB;                    // A-permuted
  half_t* kb  = qb + Mtok * EMB;                    // plain
  half_t* vpk = kb + Mtok * EMB;                    // V B-packed
  half_t* ob  = vpk + VP_HALVES;                    // A-permuted
  half_t* w1h = ob + Mtok * EMB;                    // all weights B-packed
  half_t* w2h = w1h + 768 * EMB;
  half_t* wqh = w2h + 192 * EMB;
  half_t* wkh = wqh + EMB * EMB;
  half_t* wvh = wkh + EMB * EMB;
  half_t* wph = wvh + EMB * EMB;

  k_pack_w<<<(768 * EMB + 255) / 256, 256, 0, stream>>>(W1, w1h, 768 * EMB);
  k_pack_w<<<(192 * EMB + 255) / 256, 256, 0, stream>>>(W2, w2h, 192 * EMB);
  k_pack_w<<<(EMB * EMB + 255) / 256, 256, 0, stream>>>(Wq, wqh, EMB * EMB);
  k_pack_w<<<(EMB * EMB + 255) / 256, 256, 0, stream>>>(Wk, wkh, EMB * EMB);
  k_pack_w<<<(EMB * EMB + 255) / 256, 256, 0, stream>>>(Wv, wvh, EMB * EMB);
  k_pack_w<<<(EMB * EMB + 255) / 256, 256, 0, stream>>>(Wp, wph, EMB * EMB);

  k_write_cls<<<Bn, 256, 0, stream>>>(cls1, cls2, z);

  // p1: 196 patches of 16x16 (grid 14 wide), tokens 198..393
  k_patch_gemm<<<(Bn * 196) / 16, 128, 0, stream>>>(x, w1h, b1, z, 768, 16, 14, 1, 0, 198);
  // p2r: sampled patches pid = 4s+3 of 8x8 grid (28 wide), tokens 1..196
  k_patch_gemm<<<(Bn * 196) / 16, 128, 0, stream>>>(x, w2h, b2, z, 192, 8, 28, 4, 3, 1);

  k_ln_f16<<<(int)Mtok, 256, 0, stream>>>(z, ln1_g, ln1_b, h);

  k_gemm_qkv<<<(int)(Mtok / 16), 128, 0, stream>>>(h, wqh, bq, qb, 1);
  k_gemm_qkv<<<(int)(Mtok / 16), 128, 0, stream>>>(h, wkh, bk, kb, 0);
  k_gemm_qkv<<<(int)(Mtok / 16), 128, 0, stream>>>(h, wvh, bv, vpk, 2);

  k_attn<<<Bn * 2 * 25, 32, 0, stream>>>(qb, kb, vpk, ob);

  k_proj_ln_add<<<(int)(Mtok / 16), 128, 0, stream>>>(ob, wph, bp, ln2_g, ln2_b, z);

  k_head<<<Bn, 256, 0, stream>>>(z, lnc_g, lnc_b, Wc, bc, out);
}
